// SchNetTriple_80590766342348
// MI455X (gfx1250) — compile-verified, matrix operand order checked by reading, not run on hardware
//
#include <hip/hip_runtime.h>
#include <hip/hip_bf16.h>

#define DEVI __device__ __forceinline__

typedef __attribute__((ext_vector_type(16))) __bf16 v16bf;
typedef __attribute__((ext_vector_type(8)))  float  v8f;
typedef unsigned int u32;

// ---------------- problem constants ----------------
constexpr int B_ = 2, A_ = 48, NBH = 47, T_ = 1081;
constexpr int C_ = 128, F_ = 128, G_ = 25, L_ = 3;
constexpr int NG   = B_ * A_;    // 96 (b,a) groups
constexpr int Tp   = 1088;       // T padded: 17 * 64
constexpr int NBHp = 48;         // NBH padded to 3*16
constexpr int KT   = 64;         // 2*G = 50 padded
constexpr int KD   = 32;         // G = 25 padded
constexpr float CUT = 5.0f;
constexpr float LN2F = 0.69314718055994530942f;
constexpr float PI_OVER_CUT = 0.62831853071795864769f;

// ---------------- workspace layout (bytes, all 16B aligned) ----------------
constexpr size_t OFF_DIJK = 0;                                   // NG*Tp*KT bf16
constexpr size_t OFF_CUTT = OFF_DIJK + (size_t)NG * Tp * KT * 2; // NG*Tp f32
constexpr size_t OFF_FDBF = OFF_CUTT + (size_t)NG * Tp * 4;      // NG*NBHp*KD bf16
constexpr size_t OFF_CUTD = OFF_FDBF + (size_t)NG * NBHp * KD * 2; // NG*NBHp f32
constexpr size_t OFF_FTW1 = OFF_CUTD + (size_t)NG * NBHp * 4;    // L*F*KT bf16 (transposed)
constexpr size_t OFF_FTW2 = OFF_FTW1 + (size_t)L_ * F_ * KT * 2; // L*F*F bf16
constexpr size_t OFF_FDW1 = OFF_FTW2 + (size_t)L_ * F_ * F_ * 2; // L*F*KD bf16
constexpr size_t OFF_FDW2 = OFF_FDW1 + (size_t)L_ * F_ * KD * 2; // L*F*F
constexpr size_t OFF_IN2F = OFF_FDW2 + (size_t)L_ * F_ * F_ * 2; // L*F*F
constexpr size_t OFF_F2O  = OFF_IN2F + (size_t)L_ * F_ * F_ * 2; // L*C*F
constexpr size_t OFF_DEN  = OFF_F2O  + (size_t)L_ * F_ * F_ * 2; // L*C*C
constexpr size_t OFF_Y    = OFF_DEN  + (size_t)L_ * F_ * F_ * 2; // NG*F f32
constexpr size_t OFF_VACC = OFF_Y    + (size_t)NG * F_ * 4;      // NG*F f32

// ---------------- helpers ----------------
union FragU { u32 u[8]; v16bf v; };

DEVI v8f wmma_bf16(v16bf a, v16bf b, v8f c) {
  return __builtin_amdgcn_wmma_f32_16x16x32_bf16(false, a, false, b, (short)0, c,
                                                 false, false);
}

// Load a 16x32 bf16 fragment (A layout; B stored transposed uses the same map).
// lane l: row/col = mn + (l&15); VGPR v holds K pair k = 16*(v/4) + 8*(l>>4) + 2*(v%4)
DEVI v16bf ldfrag(const __bf16* base, int ld, int mn, int kb) {
  const int lane = (int)(threadIdx.x & 31u);
  const __bf16* p = base + (mn + (lane & 15)) * ld + kb + ((lane >> 4) << 3);
  FragU f;
  const u32* a = (const u32*)p;
  const u32* b = (const u32*)(p + 16);
  f.u[0] = a[0]; f.u[1] = a[1]; f.u[2] = a[2]; f.u[3] = a[3];
  f.u[4] = b[0]; f.u[5] = b[1]; f.u[6] = b[2]; f.u[7] = b[3];
  return f.v;
}

// ---- gfx1250 async global->LDS copy (ASYNCcnt path) ----
// 16B per lane: LDS byte offset in VDST vgpr, global = SADDR(base) + VADDR(off).
DEVI void async_cp16(void* lds_ptr, const void* gbase, u32 gbyte_off) {
  u32 lds_off = (u32)(uintptr_t)lds_ptr;  // low 32 bits of LDS aperture = LDS addr
  asm volatile("global_load_async_to_lds_b128 %0, %1, %2"
               :: "v"(lds_off), "v"(gbyte_off), "s"(gbase)
               : "memory");
}
DEVI void async_wait0() {
#if __has_builtin(__builtin_amdgcn_s_wait_asynccnt)
  __builtin_amdgcn_s_wait_asynccnt(0);
#else
  asm volatile("s_wait_asynccnt 0x0" ::: "memory");
#endif
}

DEVI float sspf(float x) {  // softplus(x) - ln2, numerically stable
  return fmaxf(x, 0.0f) + log1pf(expf(-fabsf(x))) - LN2F;
}
DEVI float ccut(float r) {  // cosine cutoff
  float c = 0.5f * (cosf(PI_OVER_CUT * r) + 1.0f);
  return (r < CUT) ? c : 0.0f;
}
DEVI float sdist(const float* p, const float* q) {
  float dx = p[0] - q[0], dy = p[1] - q[1], dz = p[2] - q[2];
  return sqrtf(fmaxf(dx * dx + dy * dy + dz * dz, 1e-12f));
}

// ---------------- precompute / utility kernels ----------------

// transpose + convert: src fp32 [K,N] -> dst bf16 [N,Kp] (K..Kp-1 zero padded)
__global__ void k_convt(const float* __restrict__ src, __bf16* __restrict__ dst,
                        int K, int N, int Kp) {
  int i = blockIdx.x * 256 + threadIdx.x;
  if (i >= N * Kp) return;
  int n = i / Kp, k = i % Kp;
  float v = (k < K) ? src[k * N + n] : 0.0f;
  dst[(size_t)n * Kp + k] = (__bf16)v;
}

__global__ void k_zero(float* p, int n) {
  int i = blockIdx.x * 256 + threadIdx.x;
  if (i < n) p[i] = 0.0f;
}

__global__ void k_embed(const int* __restrict__ an, const float* __restrict__ emb,
                        float* __restrict__ x) {
  int i = blockIdx.x * 256 + threadIdx.x;
  if (i >= NG * C_) return;
  int g = i / C_, c = i % C_;
  x[i] = emb[an[g] * C_ + c];
}

__global__ void k_pair(const float* __restrict__ pos, const int* __restrict__ nbr,
                       const float* __restrict__ nmask, __bf16* __restrict__ fdbf,
                       float* __restrict__ cutd) {
  int i = blockIdx.x * 256 + threadIdx.x;
  if (i >= NG * NBHp) return;
  int g = i / NBHp, n = i % NBHp;
  int b = g / A_, a = g % A_;
  float r = 0.0f, m = 0.0f;
  if (n < NBH) {
    int j = nbr[g * NBH + n];
    m = nmask[g * NBH + n];
    r = sdist(pos + (b * A_ + j) * 3, pos + (b * A_ + a) * 3) * m;
  }
  cutd[i] = ccut(r) * m;
  const float step = CUT / (G_ - 1);
  const float coeff = -0.5f / (step * step);
  __bf16* row = fdbf + (size_t)i * KD;
  for (int gg = 0; gg < KD; ++gg) {
    float f = 0.0f;
    if (n < NBH && gg < G_) {
      float d = r - gg * step;
      f = expf(coeff * d * d) * m;
    }
    row[gg] = (__bf16)f;
  }
}

__global__ void k_triple(const float* __restrict__ pos, const int* __restrict__ nbrj,
                         const int* __restrict__ nbrk, const float* __restrict__ tmask,
                         __bf16* __restrict__ dijk, float* __restrict__ cutt) {
  int i = blockIdx.x * 256 + threadIdx.x;
  if (i >= NG * Tp) return;
  int g = i / Tp, t = i % Tp;
  int b = g / A_, a = g % A_;
  __bf16* row = dijk + (size_t)i * KT;
  if (t >= T_) {
    cutt[i] = 0.0f;
    for (int k = 0; k < KT; ++k) row[k] = (__bf16)0.0f;
    return;
  }
  int j  = nbrj[(size_t)g * T_ + t];
  int kk = nbrk[(size_t)g * T_ + t];
  float m = tmask[(size_t)g * T_ + t];
  const float* pi = pos + (b * A_ + a) * 3;
  const float* pj = pos + (b * A_ + j) * 3;
  const float* pk = pos + (b * A_ + kk) * 3;
  float r_ij = sdist(pj, pi) * m;
  float r_ik = sdist(pk, pi) * m;
  float r_jk = sdist(pk, pj) * m;
  float cosv = (r_ij * r_ij + r_ik * r_ik - r_jk * r_jk) /
               (2.0f * r_ij * r_ik + 1e-10f);
  cosv = fminf(1.0f, fmaxf(-1.0f, cosv));
  float ang = 1.0f + cosv;  // NZ==1: 2^(1-1)*(1+cos)^1
  cutt[i] = ccut(r_ij) * ccut(r_jk) * m;
  const float step = CUT / (G_ - 1);
  const float coeff = -0.5f / (step * step);
  for (int gg = 0; gg < G_; ++gg) {
    float d1 = r_ij - gg * step;
    float d2 = r_jk - gg * step;
    float fij = expf(coeff * d1 * d1) * m;   // gaussian * triple_mask
    float fjk = expf(coeff * d2 * d2) * m;
    row[gg]      = (__bf16)(ang * fij * m);  // outer *triple_mask as in reference
    row[G_ + gg] = (__bf16)(ang * fjk * m);
  }
  for (int k = 2 * G_; k < KT; ++k) row[k] = (__bf16)0.0f;
}

// ---------------- y = x @ in2f_w[l] : one block, WMMA ----------------
__global__ __launch_bounds__(256) void k_y(const float* __restrict__ x,
                                           const __bf16* __restrict__ wt,
                                           float* __restrict__ y) {
  __shared__ __align__(16) char smem[96 * 128 * 2 + 128 * 128 * 2];
  __bf16* lX = (__bf16*)smem;
  __bf16* lB = (__bf16*)(smem + 96 * 128 * 2);
  // weights: async copy engine; x: fp32->bf16 conversion on VALU path
  for (int i = threadIdx.x; i < 2048; i += 256)
    async_cp16((char*)lB + i * 16, wt, (u32)(i * 16));
  for (int i = threadIdx.x; i < 96 * 128; i += 256) lX[i] = (__bf16)x[i];
  async_wait0();
  __syncthreads();
  int w = threadIdx.x >> 5, lane = threadIdx.x & 31;
  int cb = w * 16;
  for (int rt = 0; rt < 6; ++rt) {
    v8f acc = {};
#pragma unroll
    for (int ks = 0; ks < 128; ks += 32)
      acc = wmma_bf16(ldfrag(lX, 128, rt * 16, ks), ldfrag(lB, 128, cb, ks), acc);
    int col = cb + (lane & 15);
    int rh = (lane >> 4) << 3;
#pragma unroll
    for (int p = 0; p < 8; ++p) y[(rt * 16 + rh + p) * 128 + col] = acc[p];
  }
}

// ---------------- fused pairwise filter + v_d reduction ----------------
// one block per (b,a) group: rows 48 (padded NBH), MLP 32->128->128 via WMMA
__global__ __launch_bounds__(256) void k_wd(const __bf16* __restrict__ fdbf,
                                            const float* __restrict__ cutd,
                                            const int* __restrict__ nbr,
                                            const __bf16* __restrict__ w1t,
                                            const __bf16* __restrict__ w2t,
                                            const float* __restrict__ b1,
                                            const float* __restrict__ b2,
                                            const float* __restrict__ y,
                                            float* __restrict__ vacc) {
  constexpr int R0 = 48 * 128 * 2;  // 12288: lH overlays lA(3072)+lB1(8192)
  __shared__ __align__(16) char smem[R0 + 128 * 128 * 2 + 48 * 4 + 48 * 4 + 128 * 4];
  __bf16* lA  = (__bf16*)smem;
  __bf16* lB1 = (__bf16*)(smem + 3072);
  __bf16* lH  = (__bf16*)smem;
  __bf16* lB2 = (__bf16*)(smem + R0);
  float* rowCut = (float*)(smem + R0 + 32768);
  int*   rowN   = (int*)(smem + R0 + 32768 + 192);
  float* lRed   = (float*)(smem + R0 + 32768 + 384);

  int g = blockIdx.x, b = g / A_;
  int tid = threadIdx.x;
  const char* gA = (const char*)(fdbf + (size_t)g * NBHp * KD);
  for (int i = tid; i < 192; i += 256)                       // 48*32 bf16 = 3 KB
    async_cp16((char*)lA + i * 16, gA, (u32)(i * 16));
  for (int i = tid; i < 512; i += 256)                       // 128*32 bf16 = 8 KB
    async_cp16((char*)lB1 + i * 16, w1t, (u32)(i * 16));
  for (int i = tid; i < 2048; i += 256)                      // 128*128 bf16 = 32 KB
    async_cp16((char*)lB2 + i * 16, w2t, (u32)(i * 16));
  if (tid < 48) {
    rowCut[tid] = cutd[g * NBHp + tid];
    rowN[tid]   = (tid < NBH) ? nbr[g * NBH + tid] : 0;
  }
  if (tid < 128) lRed[tid] = 0.0f;
  async_wait0();
  __syncthreads();

  int w = tid >> 5, lane = tid & 31;
  int cb = w * 16;
  v8f acc[3];
#pragma unroll
  for (int rt = 0; rt < 3; ++rt) {
    v8f z = {};
    acc[rt] = wmma_bf16(ldfrag(lA, KD, rt * 16, 0), ldfrag(lB1, KD, cb, 0), z);
  }
  __syncthreads();  // all waves done reading lA/lB1
  {
    int col = cb + (lane & 15), rh = (lane >> 4) << 3;
    float bb1 = b1[col];
#pragma unroll
    for (int rt = 0; rt < 3; ++rt)
#pragma unroll
      for (int p = 0; p < 8; ++p)
        lH[(rt * 16 + rh + p) * 128 + col] = (__bf16)sspf(acc[rt][p] + bb1);
  }
  __syncthreads();
#pragma unroll
  for (int rt = 0; rt < 3; ++rt) {
    v8f a2 = {};
#pragma unroll
    for (int ks = 0; ks < 128; ks += 32)
      a2 = wmma_bf16(ldfrag(lH, 128, rt * 16, ks), ldfrag(lB2, 128, cb, ks), a2);
    int col = cb + (lane & 15), rh = (lane >> 4) << 3;
    float bb2 = b2[col];
    float s = 0.0f;
#pragma unroll
    for (int p = 0; p < 8; ++p) {
      int row = rt * 16 + rh + p;
      float wd = sspf(a2[p] + bb2) * rowCut[row];
      s += wd * y[(b * A_ + rowN[row]) * 128 + col];
    }
    atomicAdd(&lRed[col], s);
  }
  __syncthreads();
  if (tid < 128) atomicAdd(&vacc[g * 128 + tid], lRed[tid]);
}

// ---------------- fused triple filter + v_t reduction ----------------
// one block per 64 triple-rows (17 blocks per (b,a) group): MLP 64->128->128
__global__ __launch_bounds__(256) void k_wt(const __bf16* __restrict__ dijk,
                                            const float* __restrict__ cutt,
                                            const int* __restrict__ nbrj,
                                            const int* __restrict__ nbrk,
                                            const __bf16* __restrict__ w1t,
                                            const __bf16* __restrict__ w2t,
                                            const float* __restrict__ b1,
                                            const float* __restrict__ b2,
                                            const float* __restrict__ y,
                                            float* __restrict__ vacc) {
  constexpr int R0 = 24576;  // lA(8192)+lB1(16384); lH(16384) overlays
  __shared__ __align__(16) char smem[R0 + 32768 + 3 * 256 + 512];
  __bf16* lA  = (__bf16*)smem;
  __bf16* lB1 = (__bf16*)(smem + 8192);
  __bf16* lH  = (__bf16*)smem;
  __bf16* lB2 = (__bf16*)(smem + R0);
  float* rowCut = (float*)(smem + R0 + 32768);
  int*   rowJ   = (int*)(smem + R0 + 32768 + 256);
  int*   rowK   = (int*)(smem + R0 + 32768 + 512);
  float* lRed   = (float*)(smem + R0 + 32768 + 768);

  int g = blockIdx.x / 17, chunk = blockIdx.x % 17;
  int t0 = chunk * 64;
  int b = g / A_;
  int tid = threadIdx.x;

  // async global->LDS staging (all bf16, no conversion needed)
  const char* gA = (const char*)(dijk + ((size_t)g * Tp + t0) * KT);
  for (int i = tid; i < 512; i += 256)                       // 64x64 bf16 = 8 KB
    async_cp16((char*)lA + i * 16, gA, (u32)(i * 16));
  for (int i = tid; i < 1024; i += 256)                      // 128x64 bf16 = 16 KB
    async_cp16((char*)lB1 + i * 16, w1t, (u32)(i * 16));
  for (int i = tid; i < 2048; i += 256)                      // 128x128 bf16 = 32 KB
    async_cp16((char*)lB2 + i * 16, w2t, (u32)(i * 16));
  if (tid < 64) {
    int t = t0 + tid;
    rowCut[tid] = cutt[(size_t)g * Tp + t];
    rowJ[tid] = (t < T_) ? nbrj[(size_t)g * T_ + t] : 0;
    rowK[tid] = (t < T_) ? nbrk[(size_t)g * T_ + t] : 0;
  }
  if (tid < 128) lRed[tid] = 0.0f;
  async_wait0();
  __syncthreads();

  int w = tid >> 5, lane = tid & 31;
  int rbase = (w >> 1) * 16;      // 4 row tiles over 8 waves
  int chb   = (w & 1) * 64;       // each wave owns 4 col tiles
  v8f acc0 = {}, acc1 = {}, acc2 = {}, acc3 = {};
#pragma unroll
  for (int ks = 0; ks < KT; ks += 32) {
    v16bf af = ldfrag(lA, KT, rbase, ks);
    acc0 = wmma_bf16(af, ldfrag(lB1, KT, chb +  0, ks), acc0);
    acc1 = wmma_bf16(af, ldfrag(lB1, KT, chb + 16, ks), acc1);
    acc2 = wmma_bf16(af, ldfrag(lB1, KT, chb + 32, ks), acc2);
    acc3 = wmma_bf16(af, ldfrag(lB1, KT, chb + 48, ks), acc3);
  }
  __syncthreads();  // GEMM1 reads complete before lH overlays lA/lB1
  {
    int c0 = chb + (lane & 15), rh = (lane >> 4) << 3;
    float b10 = b1[c0], b11 = b1[c0 + 16], b12 = b1[c0 + 32], b13 = b1[c0 + 48];
#pragma unroll
    for (int p = 0; p < 8; ++p) {
      int row = rbase + rh + p;
      lH[row * 128 + c0 +  0] = (__bf16)sspf(acc0[p] + b10);
      lH[row * 128 + c0 + 16] = (__bf16)sspf(acc1[p] + b11);
      lH[row * 128 + c0 + 32] = (__bf16)sspf(acc2[p] + b12);
      lH[row * 128 + c0 + 48] = (__bf16)sspf(acc3[p] + b13);
    }
  }
  __syncthreads();
#pragma unroll
  for (int ct = 0; ct < 4; ++ct) {
    int cb = chb + ct * 16;
    v8f a2 = {};
#pragma unroll
    for (int ks = 0; ks < 128; ks += 32)
      a2 = wmma_bf16(ldfrag(lH, 128, rbase, ks), ldfrag(lB2, 128, cb, ks), a2);
    int col = cb + (lane & 15), rh = (lane >> 4) << 3;
    float bb2 = b2[col];
    float s = 0.0f;
#pragma unroll
    for (int p = 0; p < 8; ++p) {
      int row = rbase + rh + p;
      float wt = sspf(a2[p] + bb2) * rowCut[row];
      float yj = y[(b * A_ + rowJ[row]) * 128 + col];
      float yk = y[(b * A_ + rowK[row]) * 128 + col];
      s += wt * yj * yk;
    }
    atomicAdd(&lRed[col], s);
  }
  __syncthreads();
  if (tid < 128) atomicAdd(&vacc[g * 128 + tid], lRed[tid]);
}

// ---------------- head: x += ssp(v@f2out+b)@dense + b ----------------
__global__ __launch_bounds__(256) void k_head(const float* __restrict__ vacc,
                                              const __bf16* __restrict__ f2ot,
                                              const __bf16* __restrict__ dent,
                                              const float* __restrict__ f2ob,
                                              const float* __restrict__ denb,
                                              float* __restrict__ x) {
  __shared__ __align__(16) char smem[96 * 128 * 2 + 128 * 128 * 2];
  __bf16* lV = (__bf16*)smem;  // then reused as lH
  __bf16* lW = (__bf16*)(smem + 96 * 128 * 2);
  int tid = threadIdx.x;
  for (int i = tid; i < 2048; i += 256)
    async_cp16((char*)lW + i * 16, f2ot, (u32)(i * 16));
  for (int i = tid; i < 96 * 128; i += 256) lV[i] = (__bf16)vacc[i];
  async_wait0();
  __syncthreads();
  int w = tid >> 5, lane = tid & 31;
  int cb = w * 16;
  v8f acc[6];
#pragma unroll
  for (int rt = 0; rt < 6; ++rt) {
    v8f a = {};
#pragma unroll
    for (int ks = 0; ks < 128; ks += 32)
      a = wmma_bf16(ldfrag(lV, 128, rt * 16, ks), ldfrag(lW, 128, cb, ks), a);
    acc[rt] = a;
  }
  __syncthreads();  // GEMM1 done; safe to overwrite lV and lW
  {
    int col = cb + (lane & 15), rh = (lane >> 4) << 3;
    float bb = f2ob[col];
#pragma unroll
    for (int rt = 0; rt < 6; ++rt)
#pragma unroll
      for (int p = 0; p < 8; ++p)
        lV[(rt * 16 + rh + p) * 128 + col] = (__bf16)sspf(acc[rt][p] + bb);
  }
  for (int i = tid; i < 2048; i += 256)
    async_cp16((char*)lW + i * 16, dent, (u32)(i * 16));
  async_wait0();
  __syncthreads();
#pragma unroll
  for (int rt = 0; rt < 6; ++rt) {
    v8f a = {};
#pragma unroll
    for (int ks = 0; ks < 128; ks += 32)
      a = wmma_bf16(ldfrag(lV, 128, rt * 16, ks), ldfrag(lW, 128, cb, ks), a);
    int col = cb + (lane & 15), rh = (lane >> 4) << 3;
    float bb = denb[col];
#pragma unroll
    for (int p = 0; p < 8; ++p) {
      int row = rt * 16 + rh + p;
      x[row * 128 + col] += a[p] + bb;
    }
  }
}

// ---------------- host launcher ----------------
extern "C" void kernel_launch(void* const* d_in, const int* in_sizes, int n_in,
                              void* d_out, int out_size, void* d_ws, size_t ws_size,
                              hipStream_t stream) {
  (void)in_sizes; (void)n_in; (void)out_size; (void)ws_size;
  const int*   an    = (const int*)d_in[0];
  const float* pos   = (const float*)d_in[1];
  const int*   nbr   = (const int*)d_in[2];
  const float* nmask = (const float*)d_in[3];
  const int*   nbrj  = (const int*)d_in[4];
  const int*   nbrk  = (const int*)d_in[5];
  const float* tmask = (const float*)d_in[6];
  const float* emb   = (const float*)d_in[7];
  const float* fd_w1 = (const float*)d_in[8];
  const float* fd_b1 = (const float*)d_in[9];
  const float* fd_w2 = (const float*)d_in[10];
  const float* fd_b2 = (const float*)d_in[11];
  const float* ft_w1 = (const float*)d_in[12];
  const float* ft_b1 = (const float*)d_in[13];
  const float* ft_w2 = (const float*)d_in[14];
  const float* ft_b2 = (const float*)d_in[15];
  const float* in2f  = (const float*)d_in[16];
  const float* f2o   = (const float*)d_in[17];
  const float* f2ob  = (const float*)d_in[18];
  const float* den   = (const float*)d_in[19];
  const float* denb  = (const float*)d_in[20];

  char* ws = (char*)d_ws;
  __bf16* dijk  = (__bf16*)(ws + OFF_DIJK);
  float*  cutt  = (float*)(ws + OFF_CUTT);
  __bf16* fdbf  = (__bf16*)(ws + OFF_FDBF);
  float*  cutd  = (float*)(ws + OFF_CUTD);
  __bf16* ftw1t = (__bf16*)(ws + OFF_FTW1);
  __bf16* ftw2t = (__bf16*)(ws + OFF_FTW2);
  __bf16* fdw1t = (__bf16*)(ws + OFF_FDW1);
  __bf16* fdw2t = (__bf16*)(ws + OFF_FDW2);
  __bf16* in2ft = (__bf16*)(ws + OFF_IN2F);
  __bf16* f2ot  = (__bf16*)(ws + OFF_F2O);
  __bf16* dent  = (__bf16*)(ws + OFF_DEN);
  float*  yb    = (float*)(ws + OFF_Y);
  float*  vacc  = (float*)(ws + OFF_VACC);

  auto conv = [&](const float* s, __bf16* d, int K, int N, int Kp) {
    int n = N * Kp;
    k_convt<<<(n + 255) / 256, 256, 0, stream>>>(s, d, K, N, Kp);
  };
  for (int l = 0; l < L_; ++l) {
    conv(ft_w1 + (size_t)l * 2 * G_ * F_, ftw1t + (size_t)l * F_ * KT, 2 * G_, F_, KT);
    conv(ft_w2 + (size_t)l * F_ * F_,     ftw2t + (size_t)l * F_ * F_, F_, F_, F_);
    conv(fd_w1 + (size_t)l * G_ * F_,     fdw1t + (size_t)l * F_ * KD, G_, F_, KD);
    conv(fd_w2 + (size_t)l * F_ * F_,     fdw2t + (size_t)l * F_ * F_, F_, F_, F_);
    conv(in2f  + (size_t)l * C_ * F_,     in2ft + (size_t)l * F_ * C_, C_, F_, F_);
    conv(f2o   + (size_t)l * F_ * C_,     f2ot  + (size_t)l * C_ * F_, F_, C_, C_);
    conv(den   + (size_t)l * C_ * C_,     dent  + (size_t)l * C_ * C_, C_, C_, C_);
  }

  k_pair<<<(NG * NBHp + 255) / 256, 256, 0, stream>>>(pos, nbr, nmask, fdbf, cutd);
  k_triple<<<(NG * Tp + 255) / 256, 256, 0, stream>>>(pos, nbrj, nbrk, tmask, dijk, cutt);
  k_embed<<<(NG * C_ + 255) / 256, 256, 0, stream>>>(an, emb, (float*)d_out);

  for (int l = 0; l < L_; ++l) {
    k_y<<<1, 256, 0, stream>>>((const float*)d_out, in2ft + (size_t)l * F_ * C_, yb);
    k_zero<<<(NG * F_ + 255) / 256, 256, 0, stream>>>(vacc, NG * F_);
    k_wd<<<NG, 256, 0, stream>>>(fdbf, cutd, nbr,
                                 fdw1t + (size_t)l * F_ * KD, fdw2t + (size_t)l * F_ * F_,
                                 fd_b1 + (size_t)l * F_, fd_b2 + (size_t)l * F_, yb, vacc);
    k_wt<<<NG * 17, 256, 0, stream>>>(dijk, cutt, nbrj, nbrk,
                                      ftw1t + (size_t)l * F_ * KT, ftw2t + (size_t)l * F_ * F_,
                                      ft_b1 + (size_t)l * F_, ft_b2 + (size_t)l * F_, yb, vacc);
    k_head<<<1, 256, 0, stream>>>(vacc, f2ot + (size_t)l * C_ * F_,
                                  dent + (size_t)l * C_ * C_,
                                  f2ob + (size_t)l * C_, denb + (size_t)l * C_,
                                  (float*)d_out);
  }
}
